// Encoder_58334245814633
// MI455X (gfx1250) — matrix operand order repaired and checked by reading, not compiled, not run on hardware
//
#include <hip/hip_runtime.h>

// ---------------------------------------------------------------------------
// Types for CDNA5 WMMA (wave32): 16x16x32 bf16 -> f32
// ---------------------------------------------------------------------------
typedef __attribute__((ext_vector_type(16))) __bf16 v16bf;
typedef __attribute__((ext_vector_type(8)))  float  v8f;

union V16U { uint4 u[2]; v16bf v; };

// fp32 -> bf16 (round-to-nearest-even) done with integer ops (no __bf16 math),
// returning the bf16 bits and the exactly-representable float it rounds to.
__device__ __forceinline__ __bf16 cvt_bf16_rne(float f, float* back) {
  unsigned u = __float_as_uint(f);
  unsigned r = (u + 0x7FFFu + ((u >> 16) & 1u)) & 0xFFFF0000u;
  *back = __uint_as_float(r);
  unsigned short s = (unsigned short)(r >> 16);
  return __builtin_bit_cast(__bf16, s);
}

__device__ __forceinline__ void split_bf16(float f, __bf16& hi, __bf16& lo) {
  float fh, dummy;
  hi = cvt_bf16_rne(f, &fh);
  lo = cvt_bf16_rne(f - fh, &dummy);
}

// ---------------------------------------------------------------------------
// Small elementwise kernels
// ---------------------------------------------------------------------------
__global__ void fill_kernel(float* __restrict__ p, float v, int n) {
  int i = blockIdx.x * blockDim.x + threadIdx.x;
  if (i < n) p[i] = v;
}

__global__ void deg_kernel(const float* __restrict__ ew, const int* __restrict__ col,
                           float* __restrict__ deg, int E) {
  int e = blockIdx.x * blockDim.x + threadIdx.x;
  if (e < E) atomicAdd(&deg[col[e]], ew[e]);
}

__global__ void rsqrt_kernel(float* __restrict__ d, int n) {
  int i = blockIdx.x * blockDim.x + threadIdx.x;
  if (i < n) {
    float v = d[i];
    d[i] = v > 0.f ? rsqrtf(v) : 0.f;
  }
}

__global__ void wsplit_kernel(const float* __restrict__ W,
                              __bf16* __restrict__ Wh, __bf16* __restrict__ Wl, int n) {
  int i = blockIdx.x * blockDim.x + threadIdx.x;
  if (i < n) {
    __bf16 h, l;
    split_bf16(W[i], h, l);
    Wh[i] = h; Wl[i] = l;
  }
}

// ---------------------------------------------------------------------------
// Tiled GEMM:  Hout[M x N] = X[M x K] * W[K x N]   (N = 256, K % 32 == 0)
// bf16 hi/lo split, fp32 accumulate via v_wmma_f32_16x16x32_bf16.
// Workgroup tile 128(M) x 64(N); 8 waves = 4(M) x 2(N); wave tile 32x32.
// ---------------------------------------------------------------------------
#define TM 128
#define TN 64
#define TK 32

__global__ __launch_bounds__(256) void gemm_bf16split_kernel(
    const float* __restrict__ X,
    const __bf16* __restrict__ Wh, const __bf16* __restrict__ Wl,
    float* __restrict__ Hout, int M, int K, int N) {
  __shared__ __align__(16) __bf16 Ah[TM * TK];
  __shared__ __align__(16) __bf16 Al[TM * TK];
  __shared__ __align__(16) __bf16 Bh[TN * TK];   // transposed: [col][k]
  __shared__ __align__(16) __bf16 Bl[TN * TK];

  const int tid   = threadIdx.x;
  const int lane  = tid & 31;
  const int wave  = tid >> 5;
  const int waveM = wave & 3;   // 0..3
  const int waveN = wave >> 2;  // 0..1
  const int m0 = blockIdx.x * TM;
  const int n0 = blockIdx.y * TN;

  v8f acc[2][2] = {};

  for (int k0 = 0; k0 < K; k0 += TK) {
    // ---- stage A tile (128x32 fp32 -> bf16 hi/lo planes) ----
    #pragma unroll
    for (int q = 0; q < 4; ++q) {
      int id = tid + 256 * q;        // float4 id: 0..1023
      int r  = id >> 3;              // row in tile (8 float4 per row)
      int kq = (id & 7) << 2;        // k offset 0,4,...,28
      float4 f = make_float4(0.f, 0.f, 0.f, 0.f);
      int gr = m0 + r;
      if (gr < M) f = *(const float4*)(X + (size_t)gr * K + k0 + kq);
      __bf16 h, l;
      split_bf16(f.x, h, l); Ah[r*TK + kq + 0] = h; Al[r*TK + kq + 0] = l;
      split_bf16(f.y, h, l); Ah[r*TK + kq + 1] = h; Al[r*TK + kq + 1] = l;
      split_bf16(f.z, h, l); Ah[r*TK + kq + 2] = h; Al[r*TK + kq + 2] = l;
      split_bf16(f.w, h, l); Ah[r*TK + kq + 3] = h; Al[r*TK + kq + 3] = l;
    }
    // ---- stage B tile transposed: Bh[c*32 + kk] = Wh[(k0+kk)*N + n0 + c] ----
    #pragma unroll
    for (int q = 0; q < 8; ++q) {
      int id = tid + 256 * q;        // 0..2047
      int kk = id >> 6;              // 0..31
      int c  = id & 63;              // 0..63
      size_t g = (size_t)(k0 + kk) * N + n0 + c;
      Bh[c*TK + kk] = Wh[g];
      Bl[c*TK + kk] = Wl[g];
    }
    __syncthreads();

    // ---- build fragments per ISA 7.12.2 wave32 layouts ----
    // A (16-bit, 16x32): lane<16 row M=lane, v0..3 = K0..7, v4..7 = K16..23;
    //                    lane>=16 same row,  v0..3 = K8..15, v4..7 = K24..31.
    v16bf afh[2], afl[2], bfh[2], bfl[2];
    #pragma unroll
    for (int t = 0; t < 2; ++t) {
      int r  = waveM * 32 + t * 16 + (lane & 15);
      int kb = (lane >> 4) << 3;     // 0 or 8
      V16U tmp;
      tmp.u[0] = *(const uint4*)(Ah + r*TK + kb);
      tmp.u[1] = *(const uint4*)(Ah + r*TK + 16 + kb);
      afh[t] = tmp.v;
      tmp.u[0] = *(const uint4*)(Al + r*TK + kb);
      tmp.u[1] = *(const uint4*)(Al + r*TK + 16 + kb);
      afl[t] = tmp.v;
    }
    // B (16-bit, 32x16): lane<16 col N=lane holds K0..7,K16..23; lane>=16 K8..15,K24..31.
    #pragma unroll
    for (int nb = 0; nb < 2; ++nb) {
      int c   = waveN * 32 + nb * 16 + (lane & 15);
      int kb2 = (lane >> 4) << 3;    // 0 or 8
      V16U tmp;
      tmp.u[0] = *(const uint4*)(Bh + c*TK + kb2);
      tmp.u[1] = *(const uint4*)(Bh + c*TK + kb2 + 16);
      bfh[nb] = tmp.v;
      tmp.u[0] = *(const uint4*)(Bl + c*TK + kb2);
      tmp.u[1] = *(const uint4*)(Bl + c*TK + kb2 + 16);
      bfl[nb] = tmp.v;
    }

    // ---- 2x2 tiles x 3 split terms = 12 WMMAs per K-chunk ----
    #pragma unroll
    for (int t = 0; t < 2; ++t) {
      #pragma unroll
      for (int nb = 0; nb < 2; ++nb) {
        acc[t][nb] = __builtin_amdgcn_wmma_f32_16x16x32_bf16(
            false, afh[t], false, bfh[nb], (short)0, acc[t][nb], false, false);
        acc[t][nb] = __builtin_amdgcn_wmma_f32_16x16x32_bf16(
            false, afh[t], false, bfl[nb], (short)0, acc[t][nb], false, false);
        acc[t][nb] = __builtin_amdgcn_wmma_f32_16x16x32_bf16(
            false, afl[t], false, bfh[nb], (short)0, acc[t][nb], false, false);
      }
    }
    __syncthreads();
  }

  // ---- store: C/D f32 layout: lane l, vgpr v -> M = v + 8*(l>>4), N = l&15 ----
  #pragma unroll
  for (int t = 0; t < 2; ++t) {
    #pragma unroll
    for (int nb = 0; nb < 2; ++nb) {
      int col = n0 + waveN * 32 + nb * 16 + (lane & 15);
      #pragma unroll
      for (int v = 0; v < 8; ++v) {
        int row = m0 + waveM * 32 + t * 16 + v + 8 * (lane >> 4);
        if (row < M) Hout[(size_t)row * N + col] = acc[t][nb][v];
      }
    }
  }
}

// ---------------------------------------------------------------------------
// Edge scatter: one wave per edge; agg[col] += h[row] * norm  (H = 256)
// ---------------------------------------------------------------------------
__global__ __launch_bounds__(256) void scatter_kernel(
    const float* __restrict__ h, const int* __restrict__ row,
    const int* __restrict__ col, const float* __restrict__ ew,
    const float* __restrict__ dinv, float* __restrict__ agg, int E) {
  int e = blockIdx.x * 8 + (threadIdx.x >> 5);
  if (e >= E) return;
  int lane = threadIdx.x & 31;
  int r = row[e], c = col[e];
  float norm = dinv[r] * ew[e] * dinv[c];
  const float* hp = h   + (size_t)r * 256;
  float*       ap = agg + (size_t)c * 256;
  #pragma unroll
  for (int j = 0; j < 8; ++j) {
    int idx = lane + j * 32;
    atomicAdd(ap + idx, hp[idx] * norm);
  }
}

// Self-loop term: agg[i] += h[i] * dinv[i]^2   (edge weight 1 for loops)
__global__ void selfloop_kernel(const float* __restrict__ h,
                                const float* __restrict__ dinv,
                                float* __restrict__ agg, int n, int H) {
  int idx = blockIdx.x * blockDim.x + threadIdx.x;
  if (idx >= n) return;
  int i = idx / H;
  float d = dinv[i];
  agg[idx] += h[idx] * d * d;
}

// out = (or +=) 0.5 * prelu(agg + b, a)
__global__ void finalize_kernel(const float* __restrict__ agg,
                                const float* __restrict__ b,
                                const float* __restrict__ a,
                                float* __restrict__ out, int n, int H,
                                int accumulate) {
  int idx = blockIdx.x * blockDim.x + threadIdx.x;
  if (idx >= n) return;
  int j = idx % H;
  float v = agg[idx] + b[j];
  float p = v > 0.f ? v : a[j] * v;
  p *= 0.5f;
  if (accumulate) out[idx] += p; else out[idx] = p;
}

// ---------------------------------------------------------------------------
// Host launcher
// ---------------------------------------------------------------------------
static inline int nblk(long long n, int b) { return (int)((n + b - 1) / b); }

extern "C" void kernel_launch(void* const* d_in, const int* in_sizes, int n_in,
                              void* d_out, int out_size, void* d_ws, size_t ws_size,
                              hipStream_t stream) {
  const float* x1  = (const float*)d_in[0];
  const int*   ei1 = (const int*)  d_in[1];
  const float* ew1 = (const float*)d_in[2];
  const float* x2  = (const float*)d_in[3];
  const int*   ei2 = (const int*)  d_in[4];
  const float* ew2 = (const float*)d_in[5];
  const float* W1  = (const float*)d_in[6];
  const float* b1  = (const float*)d_in[7];
  const float* W2  = (const float*)d_in[8];
  const float* b2  = (const float*)d_in[9];
  const float* a1  = (const float*)d_in[10];
  const float* a2  = (const float*)d_in[11];

  const int Hd = in_sizes[7];            // 256
  const int C1 = in_sizes[6] / Hd;       // 512
  const int C2 = in_sizes[8] / Hd;       // 256
  const int NN = in_sizes[0] / C1;       // 50000
  const int E1 = in_sizes[2];
  const int E2 = in_sizes[5];
  const long long NH = (long long)NN * Hd;

  const int* row1 = ei1;       const int* col1 = ei1 + E1;
  const int* row2 = ei2;       const int* col2 = ei2 + E2;

  // workspace layout
  char* ws = (char*)d_ws;
  size_t szH = (size_t)NN * Hd * sizeof(float);
  float*  hbuf = (float*)ws;
  float*  agg  = (float*)(ws + szH);
  float*  d1   = (float*)(ws + 2 * szH);
  float*  d2   = d1 + NN;
  __bf16* w1h  = (__bf16*)(ws + 2 * szH + 2 * (size_t)NN * sizeof(float));
  __bf16* w1l  = w1h + (size_t)C1 * Hd;
  __bf16* w2h  = w1l + (size_t)C1 * Hd;
  __bf16* w2l  = w2h + (size_t)C2 * Hd;

  // ---- normalization terms: deg = 1 (self loop) + sum_e ew at col; dinv ----
  fill_kernel<<<nblk(NN, 256), 256, 0, stream>>>(d1, 1.0f, NN);
  fill_kernel<<<nblk(NN, 256), 256, 0, stream>>>(d2, 1.0f, NN);
  deg_kernel<<<nblk(E1, 256), 256, 0, stream>>>(ew1, col1, d1, E1);
  deg_kernel<<<nblk(E2, 256), 256, 0, stream>>>(ew2, col2, d2, E2);
  rsqrt_kernel<<<nblk(NN, 256), 256, 0, stream>>>(d1, NN);
  rsqrt_kernel<<<nblk(NN, 256), 256, 0, stream>>>(d2, NN);

  // ---- split weights to bf16 hi/lo ----
  wsplit_kernel<<<nblk((long long)C1 * Hd, 256), 256, 0, stream>>>(W1, w1h, w1l, C1 * Hd);
  wsplit_kernel<<<nblk((long long)C2 * Hd, 256), 256, 0, stream>>>(W2, w2h, w2l, C2 * Hd);

  dim3 ggrid((NN + TM - 1) / TM, Hd / TN);

  // ================= layer 1 =================
  gemm_bf16split_kernel<<<ggrid, 256, 0, stream>>>(x1, w1h, w1l, hbuf, NN, C1, Hd);
  fill_kernel<<<nblk(NH, 256), 256, 0, stream>>>(agg, 0.f, (int)NH);
  scatter_kernel<<<(E1 + 7) / 8, 256, 0, stream>>>(hbuf, row1, col1, ew1, d1, agg, E1);
  selfloop_kernel<<<nblk(NH, 256), 256, 0, stream>>>(hbuf, d1, agg, (int)NH, Hd);
  finalize_kernel<<<nblk(NH, 256), 256, 0, stream>>>(agg, b1, a1, (float*)d_out,
                                                     (int)NH, Hd, 0);

  // ================= layer 2 =================
  gemm_bf16split_kernel<<<ggrid, 256, 0, stream>>>(x2, w2h, w2l, hbuf, NN, C2, Hd);
  fill_kernel<<<nblk(NH, 256), 256, 0, stream>>>(agg, 0.f, (int)NH);
  scatter_kernel<<<(E2 + 7) / 8, 256, 0, stream>>>(hbuf, row2, col2, ew2, d2, agg, E2);
  selfloop_kernel<<<nblk(NH, 256), 256, 0, stream>>>(hbuf, d2, agg, (int)NH, Hd);
  finalize_kernel<<<nblk(NH, 256), 256, 0, stream>>>(agg, b2, a2, (float*)d_out,
                                                     (int)NH, Hd, 1);
}
